// Head_63161789055255
// MI455X (gfx1250) — matrix operand order repaired and checked by reading, not compiled, not run on hardware
//
#include <hip/hip_runtime.h>
#include <stdint.h>

// Problem dims (fixed by the reference)
constexpr int BB = 4;
constexpr int T  = 2048;
constexpr int D  = 1024;
constexpr int H  = 128;
constexpr int BT = BB * T;               // 8192 rows total
constexpr float QSCALE = 0.08838834764831845f;  // 128^-0.5

typedef __attribute__((ext_vector_type(16))) __bf16 v16bf;
typedef __attribute__((ext_vector_type(8)))  float  v8f;

union AFrag {
    v16bf    v;
    uint32_t u[8];
};

__device__ __forceinline__ uint16_t f2bf(float f) {
    uint32_t u = __float_as_uint(f);
    uint32_t r = u + 0x7FFFu + ((u >> 16) & 1u);   // round-to-nearest-even
    return (uint16_t)(r >> 16);
}

__device__ __forceinline__ v8f wmma_bf16(const AFrag& a, const AFrag& b, v8f c) {
    return __builtin_amdgcn_wmma_f32_16x16x32_bf16(false, a.v, false, b.v,
                                                   (short)0, c, false, false);
}

// Async copy of one 16-byte chunk from global memory into LDS (ASYNCcnt path)
__device__ __forceinline__ void async_copy_b128(uint32_t lds_off, const void* gptr) {
    asm volatile("global_load_async_to_lds_b128 %0, %1, off"
                 :
                 : "v"(lds_off), "v"((uint64_t)(uintptr_t)gptr)
                 : "memory");
}

// ---------------------------------------------------------------------------
// Kernel 1: fp32 -> bf16 conversion of x and Wq/Wk/Wv into workspace
// ---------------------------------------------------------------------------
__global__ void cvt_kernel(const float* __restrict__ x,
                           const float* __restrict__ wq,
                           const float* __restrict__ wk,
                           const float* __restrict__ wv,
                           uint16_t* __restrict__ xb,
                           uint16_t* __restrict__ wb) {
    const int XN = BT * D;
    const int WN = H * D;
    const int total = XN + 3 * WN;
    for (int i = blockIdx.x * blockDim.x + threadIdx.x; i < total;
         i += gridDim.x * blockDim.x) {
        if (i < XN) {
            xb[i] = f2bf(x[i]);
        } else {
            int j = i - XN;
            int which = j / WN;
            int off   = j - which * WN;
            const float* w = (which == 0) ? wq : (which == 1) ? wk : wv;
            wb[j] = f2bf(w[off]);
        }
    }
}

// ---------------------------------------------------------------------------
// Kernel 2: QKV projection  out[row][h] = sum_d x[row][d] * W[h][d] + bias[h]
//   grid.x = BT/16 (16-row m-tiles), grid.y = 3 (Q,K,V)
//   256 threads = 8 waves; wave w owns h-tile w (H = 8 * 16)
//   Q pre-scaled by H^-0.5.  V stored TRANSPOSED: vt[h][row] so attention
//   B-fragments become contiguous dword loads.
// ---------------------------------------------------------------------------
__global__ __launch_bounds__(256)
void qkv_kernel(const uint16_t* __restrict__ xb,
                const uint16_t* __restrict__ wb,
                const float* __restrict__ bq,
                const float* __restrict__ bk,
                const float* __restrict__ bv,
                uint16_t* __restrict__ qb,
                uint16_t* __restrict__ kbuf,
                uint16_t* __restrict__ vt) {
    const int lane = threadIdx.x & 31;
    const int wave = threadIdx.x >> 5;
    const int n    = lane & 15;
    const int hi   = lane >> 4;
    const int sel  = blockIdx.y;
    const int mbase = blockIdx.x * 16;
    const int hbase = wave * 16;

    const uint16_t* w    = wb + (size_t)sel * H * D;
    const float*    bias = (sel == 0) ? bq : (sel == 1) ? bk : bv;

    v8f acc = {};
    for (int k0 = 0; k0 < D; k0 += 32) {
        AFrag a, bf;
        // A fragment: x rows mbase..mbase+15, K = k0..k0+31 (ISA 16-bit A layout)
        #pragma unroll
        for (int v = 0; v < 8; ++v) {
            int d = k0 + (v >> 2) * 16 + hi * 8 + (v & 3) * 2;
            a.u[v] = *(const uint32_t*)(xb + (size_t)(mbase + n) * D + d);
        }
        // B fragment: B[k][nn] = W[hbase+nn][k]  (lane = column, same K order as A)
        #pragma unroll
        for (int v = 0; v < 8; ++v) {
            int d = k0 + (v >> 2) * 16 + hi * 8 + (v & 3) * 2;
            bf.u[v] = *(const uint32_t*)(w + (size_t)(hbase + n) * D + d);
        }
        acc = wmma_bf16(a, bf, acc);
    }

    const float bval = bias[hbase + n];
    if (sel == 2) {
        // V: transposed store vt[h][row]; 8 consecutive rows per lane -> 16B run
        #pragma unroll
        for (int r = 0; r < 8; ++r) {
            int row = mbase + r + 8 * hi;
            vt[(size_t)(hbase + n) * BT + row] = f2bf(acc[r] + bval);
        }
    } else {
        uint16_t* outp = (sel == 0) ? qb : kbuf;
        const float scale = (sel == 0) ? QSCALE : 1.0f;
        #pragma unroll
        for (int r = 0; r < 8; ++r) {
            int row = mbase + r + 8 * hi;          // C/D layout: VGPR r, half hi
            outp[(size_t)row * H + hbase + n] = f2bf((acc[r] + bval) * scale);
        }
    }
}

// ---------------------------------------------------------------------------
// Kernel 3: flash attention.
//   Block = 256 threads (8 waves), each wave owns a 16-row query tile.
//   K [32][128] and V^T [128][32] tiles are DOUBLE-BUFFERED in LDS and
//   staged with global_load_async_to_lds_b128 (ASYNCcnt): while the block
//   computes on tile i, tile i+1 streams in.  Fragments come from wide DS
//   loads; P is reshaped C-layout->A-layout through per-wave LDS scratch.
// ---------------------------------------------------------------------------
constexpr int WPB   = 8;        // waves (query tiles) per block
constexpr int NSTEP = T / 32;   // 32-key steps per batch

__global__ __launch_bounds__(32 * WPB)
void attn_kernel(const uint16_t* __restrict__ qb,
                 const uint16_t* __restrict__ kbuf,
                 const uint16_t* __restrict__ vt,
                 float* __restrict__ out) {
    __shared__ __align__(16) uint16_t ktile[2][32][128];   // 2 x 8 KB
    __shared__ __align__(16) uint16_t vtile[2][128][32];   // 2 x 8 KB
    __shared__ __align__(16) uint16_t ps[WPB][16][34];     // P tiles (bf16)

    const int tid   = threadIdx.x;
    const int lane  = tid & 31;
    const int wave  = tid >> 5;
    const int n     = lane & 15;
    const int hi    = lane >> 4;
    const int qbase = (blockIdx.x * WPB + wave) * 16;
    const int batch = (blockIdx.x * WPB * 16) / T;
    const int krow0 = batch * T;

    // Stage one 32-key step (K tile + V^T tile) into LDS buffer `buf`.
    // 512 + 512 16B chunks over 256 threads -> 2 + 2 chunks per thread.
    auto stage_tiles = [&](int buf, int k0) {
        uint32_t kbase = (uint32_t)(uintptr_t)&ktile[buf][0][0];
        uint32_t vbase = (uint32_t)(uintptr_t)&vtile[buf][0][0];
        #pragma unroll
        for (int j = 0; j < 2; ++j) {
            int c = tid + 256 * j;                          // 0..511
            async_copy_b128(kbase + (uint32_t)c * 16,
                            kbuf + (size_t)(k0 + (c >> 4)) * H + (c & 15) * 8);
        }
        #pragma unroll
        for (int j = 0; j < 2; ++j) {
            int c = tid + 256 * j;                          // 0..511
            async_copy_b128(vbase + (uint32_t)c * 16,
                            vt + (size_t)(c >> 2) * BT + k0 + (c & 3) * 8);
        }
    };

    // Resident Q fragments: [16 x 128] bf16 = 4 A-fragments of K=32
    AFrag qf[4];
    #pragma unroll
    for (int db = 0; db < 4; ++db) {
        #pragma unroll
        for (int v = 0; v < 8; ++v) {
            int d = db * 32 + (v >> 2) * 16 + hi * 8 + (v & 3) * 2;
            qf[db].u[v] = *(const uint32_t*)(qb + (size_t)(qbase + n) * H + d);
        }
    }

    v8f o[8];
    float mr[8], lr[8];
    #pragma unroll
    for (int t = 0; t < 8; ++t) o[t] = {};
    #pragma unroll
    for (int r = 0; r < 8; ++r) { mr[r] = -1e30f; lr[r] = 0.0f; }

    stage_tiles(0, krow0);   // prologue: fetch step 0

    for (int it = 0; it < NSTEP; ++it) {
        const int cur = it & 1;

        asm volatile("s_wait_asynccnt 0" ::: "memory");  // tile `it` resident
        __syncthreads();  // all waves past step it-1 compute; tile it visible

        if (it + 1 < NSTEP)
            stage_tiles(cur ^ 1, krow0 + (it + 1) * 32);  // overlap next fetch

        // ---- S = Q * K^T for two 16-key tiles ----
        v8f s0 = {}, s1 = {};
        #pragma unroll
        for (int db = 0; db < 4; ++db) {
            AFrag f0, f1;
            #pragma unroll
            for (int v = 0; v < 8; ++v) {
                int d = db * 32 + (v >> 2) * 16 + hi * 8 + (v & 3) * 2;
                f0.u[v] = *(const uint32_t*)&ktile[cur][n][d];
                f1.u[v] = *(const uint32_t*)&ktile[cur][16 + n][d];
            }
            s0 = wmma_bf16(qf[db], f0, s0);
            s1 = wmma_bf16(qf[db], f1, s1);
        }

        // ---- online softmax update (per-row over 32 new keys) ----
        float p0[8], p1[8], ef[8];
        #pragma unroll
        for (int r = 0; r < 8; ++r) {
            float mx = fmaxf(s0[r], s1[r]);
            mx = fmaxf(mx, __shfl_xor(mx, 1));
            mx = fmaxf(mx, __shfl_xor(mx, 2));
            mx = fmaxf(mx, __shfl_xor(mx, 4));
            mx = fmaxf(mx, __shfl_xor(mx, 8));
            float mnew = fmaxf(mr[r], mx);
            float e0 = __expf(s0[r] - mnew);
            float e1 = __expf(s1[r] - mnew);
            float sum = e0 + e1;
            sum += __shfl_xor(sum, 1);
            sum += __shfl_xor(sum, 2);
            sum += __shfl_xor(sum, 4);
            sum += __shfl_xor(sum, 8);
            float scl = __expf(mr[r] - mnew);
            lr[r] = lr[r] * scl + sum;
            mr[r] = mnew;
            ef[r] = scl;
            p0[r] = e0;
            p1[r] = e1;
        }
        #pragma unroll
        for (int t = 0; t < 8; ++t)
            #pragma unroll
            for (int r = 0; r < 8; ++r)
                o[t][r] = o[t][r] * ef[r];

        // ---- reshape P (C layout fp32) -> A fragment (bf16) via LDS ----
        #pragma unroll
        for (int r = 0; r < 8; ++r) {
            int m = r + 8 * hi;
            ps[wave][m][n]      = f2bf(p0[r]);
            ps[wave][m][16 + n] = f2bf(p1[r]);
        }
        asm volatile("s_wait_dscnt 0" ::: "memory");  // LDS RAW within wave

        AFrag pf;
        #pragma unroll
        for (int v = 0; v < 8; ++v) {
            int k = (v >> 2) * 16 + hi * 8 + (v & 3) * 2;
            pf.u[v] = *(const uint32_t*)&ps[wave][n][k];
        }

        // ---- O += P * V over all 8 h-tiles (V^T tile: contiguous dwords) ----
        #pragma unroll
        for (int ht = 0; ht < 8; ++ht) {
            AFrag vf;
            #pragma unroll
            for (int v = 0; v < 8; ++v) {
                vf.u[v] =
                    *(const uint32_t*)&vtile[cur][ht * 16 + n][(v >> 2) * 16 + hi * 8 + (v & 3) * 2];
            }
            o[ht] = wmma_bf16(pf, vf, o[ht]);
        }
    }

    // ---- normalize and write output (fp32) ----
    #pragma unroll
    for (int r = 0; r < 8; ++r) {
        float inv = 1.0f / lr[r];
        int row = qbase + r + 8 * hi;
        #pragma unroll
        for (int ht = 0; ht < 8; ++ht)
            out[(size_t)row * H + ht * 16 + n] = o[ht][r] * inv;
    }
}

// ---------------------------------------------------------------------------
extern "C" void kernel_launch(void* const* d_in, const int* in_sizes, int n_in,
                              void* d_out, int out_size, void* d_ws, size_t ws_size,
                              hipStream_t stream) {
    const float* x  = (const float*)d_in[0];
    const float* Wq = (const float*)d_in[1];
    const float* bq = (const float*)d_in[2];
    const float* Wk = (const float*)d_in[3];
    const float* bk = (const float*)d_in[4];
    const float* Wv = (const float*)d_in[5];
    const float* bv = (const float*)d_in[6];

    // Workspace layout (bf16 halves): x | Wq,Wk,Wv | Q | K | V^T  (~24 MB)
    uint16_t* xb = (uint16_t*)d_ws;
    uint16_t* wb = xb + (size_t)BT * D;
    uint16_t* qb = wb + (size_t)3 * H * D;
    uint16_t* kb = qb + (size_t)BT * H;
    uint16_t* vb = kb + (size_t)BT * H;

    cvt_kernel<<<2048, 256, 0, stream>>>(x, Wq, Wk, Wv, xb, wb);
    qkv_kernel<<<dim3(BT / 16, 3), 256, 0, stream>>>(xb, wb, bq, bk, bv, qb, kb, vb);
    attn_kernel<<<BT / (16 * WPB), 32 * WPB, 0, stream>>>(qb, kb, vb, (float*)d_out);
}
